// SwinTransformerBlock_627065225621
// MI455X (gfx1250) — matrix operand order repaired and checked by reading, not compiled, not run on hardware
//
#include <hip/hip_runtime.h>
#include <hip/hip_bf16.h>

typedef __attribute__((ext_vector_type(16))) _Float16 v16h;
typedef __attribute__((ext_vector_type(8)))  _Float16 v8h;
typedef __attribute__((ext_vector_type(8)))  float    v8f;

#define ROWS   50176   // B*H*W = 16*56*56 = B*NW*N
#define CDIM   384
#define HIDDIM 1536
#define NHEAD  12
#define UNITS  12288   // (B*64) * NHEAD
#define SCALE  0.17677669529663689f  // 32^-0.5

// ---------------- WMMA helpers (layouts per CDNA5 ISA 7.12.2) ----------------
__device__ __forceinline__ v16h frag2(const _Float16* p0, const _Float16* p1) {
  v8h lo = *(const v8h*)p0;
  v8h hi = *(const v8h*)p1;
  return __builtin_shufflevector(lo, hi, 0,1,2,3,4,5,6,7,8,9,10,11,12,13,14,15);
}
// A fragment: A row-major [M][ld], tile (rowBase, kBase). lane m = lane&15,
// K chunks [8*half, +8) and [16+8*half, +8).
__device__ __forceinline__ v16h load_a(const _Float16* A, int ld, int rowBase, int kBase, int lane) {
  int half = lane >> 4, m = lane & 15;
  const _Float16* p = A + (size_t)(rowBase + m) * ld + kBase + 8 * half;
  return frag2(p, p + 16);
}
// B fragment from B-transposed layout BT[N][ld] (K contiguous). lane n = lane&15,
// K chunks [8*half, +8) and [16+8*half, +8).
__device__ __forceinline__ v16h load_bt(const _Float16* BT, int ld, int nBase, int kBase, int lane) {
  int half = lane >> 4, n = lane & 15;
  const _Float16* p = BT + (size_t)(nBase + n) * ld + kBase + 8 * half;
  return frag2(p, p + 16);
}
__device__ __forceinline__ v8f wmma16(v16h a, v16h b, v8f c) {
  return __builtin_amdgcn_wmma_f32_16x16x32_f16(false, a, false, b, (short)0, c, false, false);
}

// ---- pipelined 32(M) x 64(N) wave-tile GEMM core, double-buffered K loop ----
// acc[mi][j]: mi=row-block of 16, j=16-wide col block. 8 WMMAs per k-step,
// 12 b128 loads per k-step (1.5 loads/WMMA).
template <int KDIM>
__device__ __forceinline__ void gemm_core_32x64(const _Float16* __restrict__ A,
                                                const _Float16* __restrict__ BT,
                                                int rowBase, int nBase, int lane,
                                                v8f acc[2][4]) {
  v16h a0 = load_a(A, KDIM, rowBase,      0, lane);
  v16h a1 = load_a(A, KDIM, rowBase + 16, 0, lane);
  v16h b0 = load_bt(BT, KDIM, nBase,      0, lane);
  v16h b1 = load_bt(BT, KDIM, nBase + 16, 0, lane);
  v16h b2 = load_bt(BT, KDIM, nBase + 32, 0, lane);
  v16h b3 = load_bt(BT, KDIM, nBase + 48, 0, lane);
  #pragma unroll 4
  for (int kb = 0; kb < KDIM - 32; kb += 32) {
    // prefetch next k-step into fresh registers (overlaps with WMMAs below)
    v16h na0 = load_a(A, KDIM, rowBase,      kb + 32, lane);
    v16h na1 = load_a(A, KDIM, rowBase + 16, kb + 32, lane);
    v16h nb0 = load_bt(BT, KDIM, nBase,      kb + 32, lane);
    v16h nb1 = load_bt(BT, KDIM, nBase + 16, kb + 32, lane);
    v16h nb2 = load_bt(BT, KDIM, nBase + 32, kb + 32, lane);
    v16h nb3 = load_bt(BT, KDIM, nBase + 48, kb + 32, lane);
    acc[0][0] = wmma16(a0, b0, acc[0][0]);
    acc[1][0] = wmma16(a1, b0, acc[1][0]);
    acc[0][1] = wmma16(a0, b1, acc[0][1]);
    acc[1][1] = wmma16(a1, b1, acc[1][1]);
    acc[0][2] = wmma16(a0, b2, acc[0][2]);
    acc[1][2] = wmma16(a1, b2, acc[1][2]);
    acc[0][3] = wmma16(a0, b3, acc[0][3]);
    acc[1][3] = wmma16(a1, b3, acc[1][3]);
    a0 = na0; a1 = na1; b0 = nb0; b1 = nb1; b2 = nb2; b3 = nb3;
  }
  acc[0][0] = wmma16(a0, b0, acc[0][0]);
  acc[1][0] = wmma16(a1, b0, acc[1][0]);
  acc[0][1] = wmma16(a0, b1, acc[0][1]);
  acc[1][1] = wmma16(a1, b1, acc[1][1]);
  acc[0][2] = wmma16(a0, b2, acc[0][2]);
  acc[1][2] = wmma16(a1, b2, acc[1][2]);
  acc[0][3] = wmma16(a0, b3, acc[0][3]);
  acc[1][3] = wmma16(a1, b3, acc[1][3]);
}

// ---------------- weight convert + transpose to f16 [N][K] ----------------
__global__ void wconvT(const float* __restrict__ W, _Float16* __restrict__ WT, int K, int N) {
  int idx = blockIdx.x * 256 + threadIdx.x;
  if (idx >= K * N) return;
  int k = idx / N, n = idx - k * N;
  WT[(size_t)n * K + k] = (_Float16)W[idx];
}

// ---------------- LN1 + cyclic shift + window partition ----------------
__global__ __launch_bounds__(128) void ln_shift_win(const float* __restrict__ x,
                                                    const float* __restrict__ gg,
                                                    const float* __restrict__ bb,
                                                    _Float16* __restrict__ out) {
  int row = blockIdx.x;                 // b*3136 + h*56 + w
  const float* xr = x + (size_t)row * CDIM;
  int t = threadIdx.x;
  float v0 = xr[t], v1 = xr[t + 128], v2 = xr[t + 256];
  float s = v0 + v1 + v2, ss = v0 * v0 + v1 * v1 + v2 * v2;
  #pragma unroll
  for (int o = 16; o; o >>= 1) { s += __shfl_xor(s, o, 32); ss += __shfl_xor(ss, o, 32); }
  __shared__ float red[2][4];
  int wid = t >> 5;
  if ((t & 31) == 0) { red[0][wid] = s; red[1][wid] = ss; }
  __syncthreads();
  s  = red[0][0] + red[0][1] + red[0][2] + red[0][3];
  ss = red[1][0] + red[1][1] + red[1][2] + red[1][3];
  float mean = s * (1.f / CDIM);
  float rstd = rsqrtf(ss * (1.f / CDIM) - mean * mean + 1e-5f);
  int b = row / 3136, hw = row - b * 3136, h = hw / 56, w = hw - h * 56;
  int h2 = h - 3; h2 += (h2 < 0) ? 56 : 0;
  int w2 = w - 3; w2 += (w2 < 0) ? 56 : 0;
  int wi = b * 64 + (h2 / 7) * 8 + (w2 / 7);
  int tt = (h2 % 7) * 7 + (w2 % 7);
  _Float16* o = out + ((size_t)wi * 49 + tt) * CDIM;
  o[t]       = (_Float16)((v0 - mean) * rstd * gg[t]       + bb[t]);
  o[t + 128] = (_Float16)((v1 - mean) * rstd * gg[t + 128] + bb[t + 128]);
  o[t + 256] = (_Float16)((v2 - mean) * rstd * gg[t + 256] + bb[t + 256]);
}

// ---------------- plain LN (LN2) ----------------
__global__ __launch_bounds__(128) void ln_plain(const float* __restrict__ x,
                                                const float* __restrict__ gg,
                                                const float* __restrict__ bb,
                                                _Float16* __restrict__ out) {
  int row = blockIdx.x;
  const float* xr = x + (size_t)row * CDIM;
  int t = threadIdx.x;
  float v0 = xr[t], v1 = xr[t + 128], v2 = xr[t + 256];
  float s = v0 + v1 + v2, ss = v0 * v0 + v1 * v1 + v2 * v2;
  #pragma unroll
  for (int o = 16; o; o >>= 1) { s += __shfl_xor(s, o, 32); ss += __shfl_xor(ss, o, 32); }
  __shared__ float red[2][4];
  int wid = t >> 5;
  if ((t & 31) == 0) { red[0][wid] = s; red[1][wid] = ss; }
  __syncthreads();
  s  = red[0][0] + red[0][1] + red[0][2] + red[0][3];
  ss = red[1][0] + red[1][1] + red[1][2] + red[1][3];
  float mean = s * (1.f / CDIM);
  float rstd = rsqrtf(ss * (1.f / CDIM) - mean * mean + 1e-5f);
  _Float16* o = out + (size_t)row * CDIM;
  o[t]       = (_Float16)((v0 - mean) * rstd * gg[t]       + bb[t]);
  o[t + 128] = (_Float16)((v1 - mean) * rstd * gg[t + 128] + bb[t + 128]);
  o[t + 256] = (_Float16)((v2 - mean) * rstd * gg[t + 256] + bb[t + 256]);
}

// ---------------- qkv GEMM: [ROWS,384]x[384,1152], scatter to padded q/k/vT ----
__global__ __launch_bounds__(256) void gemm_qkv(const _Float16* __restrict__ A,
                                                const _Float16* __restrict__ BT,
                                                const float* __restrict__ bias,
                                                _Float16* __restrict__ qP,
                                                _Float16* __restrict__ kP,
                                                _Float16* __restrict__ vP) {
  const int NT = 1152 / 64;
  int g = blockIdx.x * 8 + (threadIdx.x >> 5);
  int mt = g / NT, nt = g - mt * NT;
  int lane = threadIdx.x & 31, half = lane >> 4, ln = lane & 15;
  int rowBase = mt * 32, nBase = nt * 64;
  v8f acc[2][4] = {};
  gemm_core_32x64<CDIM>(A, BT, rowBase, nBase, lane, acc);
  #pragma unroll
  for (int j = 0; j < 4; j++) {
    int col = nBase + 16 * j + ln;
    int s = col / 384, rem = col - s * 384, hh = rem >> 5, d = rem & 31;
    float bv = bias[col];
    #pragma unroll
    for (int mi = 0; mi < 2; mi++)
      #pragma unroll
      for (int r = 0; r < 8; r++) {
        int row = rowBase + mi * 16 + r + 8 * half;
        int wi = row / 49, t = row - wi * 49;
        float v = acc[mi][j][r] + bv;
        size_t u = (size_t)(wi * NHEAD + hh);
        if (s == 0)      qP[(u * 64 + t) * 32 + d] = (_Float16)(v * SCALE);
        else if (s == 1) kP[(u * 64 + t) * 32 + d] = (_Float16)v;
        else             vP[(u * 32 + d) * 64 + t] = (_Float16)v;  // v stored transposed
      }
  }
}

// ---------------- attention: one block(4 waves) per (window,head) -------------
__global__ __launch_bounds__(128) void attn_win(const _Float16* __restrict__ qP,
                                                const _Float16* __restrict__ kP,
                                                const _Float16* __restrict__ vP,
                                                const float* __restrict__ rpb_table,
                                                const int* __restrict__ rel_idx,
                                                const float* __restrict__ attn_mask,
                                                _Float16* __restrict__ aout) {
  int unit = blockIdx.x;                    // wi*12 + hh
  int wi = unit / NHEAD, hh = unit - wi * NHEAD;
  int wimg = wi & 63;
  int wid = threadIdx.x >> 5, lane = threadIdx.x & 31;
  int half = lane >> 4, ln = lane & 15;
  const _Float16* q  = qP + (size_t)unit * 2048;
  const _Float16* k  = kP + (size_t)unit * 2048;
  const _Float16* vt = vP + (size_t)unit * 2048;
  int rowBase = wid * 16;
  // ---- scores = (q*scale) @ k^T, 16x64 band, K=32 ----
  v8f sc[4] = {};
  {
    v16h a = load_a(q, 32, rowBase, 0, lane);
    #pragma unroll
    for (int j = 0; j < 4; j++) {
      v16h b = load_bt(k, 32, 16 * j, 0, lane);
      sc[j] = wmma16(a, b, sc[j]);
    }
  }
  // ---- bias + mask + softmax over cols 0..48 ----
  float pr[4][8], rowmax[8], rowsum[8];
  #pragma unroll
  for (int r = 0; r < 8; r++) { rowmax[r] = -1e30f; rowsum[r] = 0.f; }
  #pragma unroll
  for (int j = 0; j < 4; j++) {
    int col = 16 * j + ln;
    #pragma unroll
    for (int r = 0; r < 8; r++) {
      int row = rowBase + 8 * half + r;
      float v;
      if (row < 49 && col < 49) {
        v = sc[j][r]
          + rpb_table[rel_idx[row * 49 + col] * NHEAD + hh]
          + attn_mask[((size_t)wimg * 49 + row) * 49 + col];
      } else {
        v = -1e30f;                 // pad columns/rows never contribute
      }
      pr[j][r] = v;
      rowmax[r] = fmaxf(rowmax[r], v);
    }
  }
  #pragma unroll
  for (int o = 1; o < 16; o <<= 1)
    #pragma unroll
    for (int r = 0; r < 8; r++) rowmax[r] = fmaxf(rowmax[r], __shfl_xor(rowmax[r], o, 32));
  #pragma unroll
  for (int j = 0; j < 4; j++)
    #pragma unroll
    for (int r = 0; r < 8; r++) { pr[j][r] = __expf(pr[j][r] - rowmax[r]); rowsum[r] += pr[j][r]; }
  #pragma unroll
  for (int o = 1; o < 16; o <<= 1)
    #pragma unroll
    for (int r = 0; r < 8; r++) rowsum[r] += __shfl_xor(rowsum[r], o, 32);
  // ---- stage P (f16) in per-wave LDS slice; same wave reads it back ----
  __shared__ __attribute__((aligned(16))) _Float16 Plds[4][16][64];
  #pragma unroll
  for (int j = 0; j < 4; j++)
    #pragma unroll
    for (int r = 0; r < 8; r++)
      Plds[wid][8 * half + r][16 * j + ln] = (_Float16)(pr[j][r] / rowsum[r]);
  // ---- out = P @ v : 16x32 band, K=64 (v stored transposed [32][64]) ----
  v8f oacc[2] = {};
  #pragma unroll
  for (int kb = 0; kb < 64; kb += 32) {
    v16h a = load_a(&Plds[wid][0][0], 64, 0, kb, lane);
    #pragma unroll
    for (int j = 0; j < 2; j++) {
      v16h b = load_bt(vt, 64, 16 * j, kb, lane);
      oacc[j] = wmma16(a, b, oacc[j]);
    }
  }
  #pragma unroll
  for (int j = 0; j < 2; j++)
    #pragma unroll
    for (int r = 0; r < 8; r++) {
      int t = rowBase + 8 * half + r;
      if (t < 49)
        aout[((size_t)wi * 49 + t) * CDIM + hh * 32 + 16 * j + ln] = (_Float16)oacc[j][r];
    }
}

// ---------------- proj GEMM + window reverse + unshift + residual -------------
__global__ __launch_bounds__(256) void gemm_proj(const _Float16* __restrict__ A,
                                                 const _Float16* __restrict__ BT,
                                                 const float* __restrict__ bias,
                                                 const float* __restrict__ x,
                                                 float* __restrict__ x1) {
  const int NT = CDIM / 64;
  int g = blockIdx.x * 8 + (threadIdx.x >> 5);
  int mt = g / NT, nt = g - mt * NT;
  int lane = threadIdx.x & 31, half = lane >> 4, ln = lane & 15;
  int rowBase = mt * 32, nBase = nt * 64;
  v8f acc[2][4] = {};
  gemm_core_32x64<CDIM>(A, BT, rowBase, nBase, lane, acc);
  #pragma unroll
  for (int j = 0; j < 4; j++) {
    int col = nBase + 16 * j + ln;
    float bv = bias[col];
    #pragma unroll
    for (int mi = 0; mi < 2; mi++)
      #pragma unroll
      for (int r = 0; r < 8; r++) {
        int row = rowBase + mi * 16 + r + 8 * half;
        int wi = row / 49, t = row - wi * 49;
        int b = wi >> 6, wiw = wi & 63, wh = wiw >> 3, ww = wiw & 7;
        int i = t / 7, jj = t - i * 7;
        int h = wh * 7 + i + 3; if (h >= 56) h -= 56;
        int w = ww * 7 + jj + 3; if (w >= 56) w -= 56;
        size_t dst = ((size_t)b * 3136 + h * 56 + w) * CDIM + col;
        x1[dst] = x[dst] + acc[mi][j][r] + bv;
      }
  }
}

// ---------------- fc1 GEMM + bias + exact GELU --------------------------------
__global__ __launch_bounds__(256) void gemm_fc1(const _Float16* __restrict__ A,
                                                const _Float16* __restrict__ BT,
                                                const float* __restrict__ bias,
                                                _Float16* __restrict__ act) {
  const int NT = HIDDIM / 64;
  int g = blockIdx.x * 8 + (threadIdx.x >> 5);
  int mt = g / NT, nt = g - mt * NT;
  int lane = threadIdx.x & 31, half = lane >> 4, ln = lane & 15;
  int rowBase = mt * 32, nBase = nt * 64;
  v8f acc[2][4] = {};
  gemm_core_32x64<CDIM>(A, BT, rowBase, nBase, lane, acc);
  #pragma unroll
  for (int j = 0; j < 4; j++) {
    int col = nBase + 16 * j + ln;
    float bv = bias[col];
    #pragma unroll
    for (int mi = 0; mi < 2; mi++)
      #pragma unroll
      for (int r = 0; r < 8; r++) {
        int row = rowBase + mi * 16 + r + 8 * half;
        float v = acc[mi][j][r] + bv;
        v = 0.5f * v * (1.f + erff(v * 0.70710678118654752f));
        act[(size_t)row * HIDDIM + col] = (_Float16)v;
      }
  }
}

// ---------------- fc2 GEMM + bias + residual -> d_out -------------------------
__global__ __launch_bounds__(256) void gemm_fc2(const _Float16* __restrict__ A,
                                                const _Float16* __restrict__ BT,
                                                const float* __restrict__ bias,
                                                const float* __restrict__ x1,
                                                float* __restrict__ out) {
  const int NT = CDIM / 64;
  int g = blockIdx.x * 8 + (threadIdx.x >> 5);
  int mt = g / NT, nt = g - mt * NT;
  int lane = threadIdx.x & 31, half = lane >> 4, ln = lane & 15;
  int rowBase = mt * 32, nBase = nt * 64;
  v8f acc[2][4] = {};
  gemm_core_32x64<HIDDIM>(A, BT, rowBase, nBase, lane, acc);
  #pragma unroll
  for (int j = 0; j < 4; j++) {
    int col = nBase + 16 * j + ln;
    float bv = bias[col];
    #pragma unroll
    for (int mi = 0; mi < 2; mi++)
      #pragma unroll
      for (int r = 0; r < 8; r++) {
        int row = rowBase + mi * 16 + r + 8 * half;
        size_t idx = (size_t)row * CDIM + col;
        out[idx] = x1[idx] + acc[mi][j][r] + bv;
      }
  }
}

// ------------------------------ launcher --------------------------------------
extern "C" void kernel_launch(void* const* d_in, const int* in_sizes, int n_in,
                              void* d_out, int out_size, void* d_ws, size_t ws_size,
                              hipStream_t stream) {
  const float* x        = (const float*)d_in[0];
  const float* g1       = (const float*)d_in[1];
  const float* b1       = (const float*)d_in[2];
  const float* qkv_w    = (const float*)d_in[3];
  const float* qkv_b    = (const float*)d_in[4];
  const float* rpb      = (const float*)d_in[5];
  const float* proj_w   = (const float*)d_in[6];
  const float* proj_b   = (const float*)d_in[7];
  const float* g2       = (const float*)d_in[8];
  const float* b2       = (const float*)d_in[9];
  const float* fc1_w    = (const float*)d_in[10];
  const float* fc1_b    = (const float*)d_in[11];
  const float* fc2_w    = (const float*)d_in[12];
  const float* fc2_b    = (const float*)d_in[13];
  const int*   rel_idx  = (const int*)d_in[14];
  const float* amask    = (const float*)d_in[15];
  float* out = (float*)d_out;

  char* w = (char*)d_ws;
  size_t off = 0;
  _Float16* winH = (_Float16*)(w + off); off += (size_t)ROWS * CDIM * 2;     // 38.5 MB
  _Float16* qP   = (_Float16*)(w + off); off += (size_t)UNITS * 2048 * 2;    // 50.3 MB
  _Float16* kP   = (_Float16*)(w + off); off += (size_t)UNITS * 2048 * 2;
  _Float16* vP   = (_Float16*)(w + off); off += (size_t)UNITS * 2048 * 2;
  _Float16* aout = (_Float16*)(w + off); off += (size_t)ROWS * CDIM * 2;
  float*    x1   = (float*)   (w + off); off += (size_t)ROWS * CDIM * 4;     // 77 MB
  _Float16* h2   = (_Float16*)(w + off); off += (size_t)ROWS * CDIM * 2;
  _Float16* qkvWT= (_Float16*)(w + off); off += (size_t)1152 * 384 * 2;
  _Float16* projWT=(_Float16*)(w + off); off += (size_t)384 * 384 * 2;
  _Float16* fc1WT= (_Float16*)(w + off); off += (size_t)1536 * 384 * 2;
  _Float16* fc2WT= (_Float16*)(w + off); off += (size_t)384 * 1536 * 2;
  _Float16* act1 = (_Float16*)w;  // 154 MB, aliases winH/qP/kP/vP (consumed by then)

  // 1) weights -> f16 transposed [N][K]
  wconvT<<<(384 * 1152 + 255) / 256, 256, 0, stream>>>(qkv_w,  qkvWT, 384,  1152);
  wconvT<<<(384 * 384  + 255) / 256, 256, 0, stream>>>(proj_w, projWT, 384, 384);
  wconvT<<<(384 * 1536 + 255) / 256, 256, 0, stream>>>(fc1_w,  fc1WT, 384,  1536);
  wconvT<<<(1536 * 384 + 255) / 256, 256, 0, stream>>>(fc2_w,  fc2WT, 1536, 384);

  // 2) LN1 + shift + window partition
  ln_shift_win<<<ROWS, 128, 0, stream>>>(x, g1, b1, winH);

  // 3) qkv GEMM -> padded q/k/vT per (window,head)   (1568 M-tiles x 18 N-tiles)
  gemm_qkv<<<(1568 * 18) / 8, 256, 0, stream>>>(winH, qkvWT, qkv_b, qP, kP, vP);

  // 4) windowed attention
  attn_win<<<UNITS, 128, 0, stream>>>(qP, kP, vP, rpb, rel_idx, amask, aout);

  // 5) proj + window reverse + unshift + residual    (1568 x 6)
  gemm_proj<<<(1568 * 6) / 8, 256, 0, stream>>>(aout, projWT, proj_b, x, x1);

  // 6) LN2
  ln_plain<<<ROWS, 128, 0, stream>>>(x1, g2, b2, h2);

  // 7) fc1 + GELU                                    (1568 x 24)
  gemm_fc1<<<(1568 * 24) / 8, 256, 0, stream>>>(h2, fc1WT, fc1_b, act1);

  // 8) fc2 + residual -> output                      (1568 x 6)
  gemm_fc2<<<(1568 * 6) / 8, 256, 0, stream>>>(act1, fc2WT, fc2_b, x1, out);

  (void)in_sizes; (void)n_in; (void)out_size; (void)ws_size;
}
